// MultiHeadAttention_29515015258197
// MI455X (gfx1250) — matrix operand order repaired and checked by reading, not compiled, not run on hardware
//
#include <hip/hip_runtime.h>

// ---------------- problem constants ----------------
#define BATCH   4
#define SEQ     4096
#define DMODEL  1024
#define NHEAD   16
#define HDIM    64
#define MTOT    (BATCH*SEQ)      // 16384
#define NQKV    (3*DMODEL)       // 3072

typedef __bf16 bf16;
typedef __attribute__((ext_vector_type(16))) __bf16 bf16x16;
typedef __attribute__((ext_vector_type(8)))  float  f32x8;
typedef __attribute__((ext_vector_type(4)))  unsigned int u32x4;
typedef __attribute__((ext_vector_type(4)))  int i32x4;

union FragU { bf16x16 v; u32x4 q[2]; };

// ---- CDNA5 async global->LDS copy (guarded: falls back to vector store) ----
#if defined(__gfx1250__)
#define AS1 __attribute__((address_space(1)))
#define AS3 __attribute__((address_space(3)))
#if __has_builtin(__builtin_amdgcn_global_load_async_to_lds_b128) && \
    __has_builtin(__builtin_amdgcn_s_wait_asynccnt)
#define HAVE_ASYNC_COPY 1
#endif
#endif

__device__ inline void copy_b128_to_lds(const bf16* gsrc, bf16* ldst) {
#if defined(HAVE_ASYNC_COPY)
    __builtin_amdgcn_global_load_async_to_lds_b128((AS1 i32x4*)gsrc,
                                                   (AS3 i32x4*)ldst, 0, 0);
#else
    *(u32x4*)ldst = *(const u32x4*)gsrc;
#endif
}

__device__ inline void async_copy_wait() {
#if defined(HAVE_ASYNC_COPY)
    __builtin_amdgcn_s_wait_asynccnt(0);
#endif
}

__device__ inline f32x8 zero8() {
    f32x8 z;
    #pragma unroll
    for (int i = 0; i < 8; ++i) z[i] = 0.0f;
    return z;
}

// A-operand stripe (16x32 bf16): lane(0..15)->rows, half picks k-subsets
// {half*8..half*8+7} and {16+half*8..+7}  (ISA 7.12.2, 16-bit A 16x32)
__device__ inline bf16x16 load_fragA(const bf16* p, int ld) {
    int lane = threadIdx.x & 31;
    int row  = lane & 15;
    int half = lane >> 4;
    const bf16* r = p + row * ld + half * 8;
    FragU f;
    f.q[0] = *(const u32x4*)(r);
    f.q[1] = *(const u32x4*)(r + 16);
    return f.v;
}

// B-operand stripe (32x16 bf16) read from a K-major (row = n, col = k) tile:
// half picks k-subsets {half*8..half*8+7} and {16+half*8..+7} (matches A)
__device__ inline bf16x16 load_fragB(const bf16* p, int ld) {
    int lane = threadIdx.x & 31;
    int n    = lane & 15;
    int half = lane >> 4;
    const bf16* r = p + n * ld + half * 8;
    FragU f;
    f.q[0] = *(const u32x4*)(r);
    f.q[1] = *(const u32x4*)(r + 16);
    return f.v;
}

__device__ inline f32x8 wmma_bf16(bf16x16 a, bf16x16 b, f32x8 c) {
#if defined(__gfx1250__)
    // (neg_a, A, neg_b, B, c_mod, C, reuse_a, reuse_b)
    return __builtin_amdgcn_wmma_f32_16x16x32_bf16(false, a, false, b,
                                                   (short)0, c, false, false);
#else
    return c;
#endif
}

// ---------------- stage 1: fp32 -> bf16 ----------------
__global__ __launch_bounds__(256) void cvt_f32_bf16(const float* __restrict__ src,
                                                    bf16* __restrict__ dst,
                                                    long long n) {
    long long i = (long long)blockIdx.x * blockDim.x + threadIdx.x;
    long long stride = (long long)gridDim.x * blockDim.x;
    for (; i < n; i += stride) dst[i] = (bf16)src[i];
}

// ---------------- stages 2 & 4: bf16 WMMA GEMM ----------------
// C[M,N] = A[M,K] (row-major bf16) x W[K,N] (row-major bf16)
// mode 0: f32 output row-major [M,N]
// mode 1: scatter bf16 into q/k/v [B,H,T,HDIM]; q scaled by 1/sqrt(HDIM)
#define MBLK 128
#define NBLK 128
#define KBLK 32

__global__ __launch_bounds__(256) void gemm_bf16(const bf16* __restrict__ Amat,
                                                 const bf16* __restrict__ Wmat,
                                                 int M, int N, int K, int mode,
                                                 float* __restrict__ outF,
                                                 bf16* __restrict__ qb,
                                                 bf16* __restrict__ kb,
                                                 bf16* __restrict__ vb) {
    __shared__ bf16 lA[MBLK][KBLK];    // 8 KB, row-major [m][k]
    __shared__ bf16 lBt[NBLK][KBLK];   // 8 KB, transposed W tile: [n][k]

    const unsigned mBlocks = (unsigned)M / MBLK;
    const unsigned bm = blockIdx.x % mBlocks;
    const unsigned bn = blockIdx.x / mBlocks;
    const int tid = threadIdx.x, wave = tid >> 5, lane = tid & 31;
    const int wm = wave >> 1;          // 0..3 -> 32-row slice
    const int wn = wave & 1;           // 0..1 -> 64-col slice
    const unsigned mrow0 = bm * MBLK;
    const unsigned ncol0 = bn * NBLK;

    f32x8 acc[2][4];
    #pragma unroll
    for (int i = 0; i < 2; ++i)
        #pragma unroll
        for (int j = 0; j < 4; ++j) acc[i][j] = zero8();

    for (int k0 = 0; k0 < K; k0 += KBLK) {
        // stage A tile (128x32 bf16) as 512 x b128 chunks, async to LDS
        #pragma unroll
        for (int i = 0; i < 2; ++i) {
            int chunk = i * 256 + tid;       // 0..511
            int r = chunk >> 2;              // 0..127
            int c = (chunk & 3) * 8;         // elem col 0/8/16/24
            copy_b128_to_lds(Amat + (size_t)(mrow0 + r) * K + k0 + c, &lA[r][c]);
        }
        // stage W tile transposed via dword loads: lBt[n][k] = W[k0+k][ncol0+n]
        #pragma unroll
        for (int i = 0; i < 8; ++i) {
            int linear = i * 256 + tid;      // 0..2047 dwords
            int kk  = linear >> 6;           // 0..31
            int nn2 = linear & 63;           // dword index in row
            union { unsigned u; bf16 h[2]; } w;
            w.u = ((const unsigned*)(Wmat + (size_t)(k0 + kk) * N + ncol0))[nn2];
            lBt[nn2 * 2][kk]     = w.h[0];
            lBt[nn2 * 2 + 1][kk] = w.h[1];
        }
        async_copy_wait();
        __syncthreads();

        bf16x16 af[2], bfv[4];
        af[0] = load_fragA(&lA[wm * 32][0], KBLK);
        af[1] = load_fragA(&lA[wm * 32 + 16][0], KBLK);
        #pragma unroll
        for (int j = 0; j < 4; ++j)
            bfv[j] = load_fragB(&lBt[wn * 64 + j * 16][0], KBLK);
        #pragma unroll
        for (int i = 0; i < 2; ++i)
            #pragma unroll
            for (int j = 0; j < 4; ++j)
                acc[i][j] = wmma_bf16(af[i], bfv[j], acc[i][j]);
        __syncthreads();
    }

    // epilogue: C/D stripe -> row r + 8*half, col lane&15 (all unsigned math)
    const unsigned col = lane & 15, half = lane >> 4;
    #pragma unroll
    for (int i = 0; i < 2; ++i)
        #pragma unroll
        for (int j = 0; j < 4; ++j) {
            unsigned mbase = mrow0 + wm * 32 + i * 16;
            unsigned nbase = ncol0 + wn * 64 + j * 16;
            #pragma unroll
            for (int r = 0; r < 8; ++r) {
                unsigned m = mbase + r + 8 * half;
                unsigned n = nbase + col;
                float val = acc[i][j][r];
                if (mode == 0) {
                    outF[(size_t)m * (unsigned)N + n] = val;
                } else {
                    unsigned sel = n >> 10;          // / DMODEL
                    unsigned d   = n & 1023u;        // % DMODEL
                    unsigned h   = d >> 6, hd = d & 63u;
                    unsigned bbi = m >> 12, t = m & 4095u;
                    bf16* dst = (sel == 0) ? qb : ((sel == 1) ? kb : vb);
                    float sc  = (sel == 0) ? 0.125f : 1.0f;  // 1/sqrt(64) into Q
                    dst[(((size_t)bbi * NHEAD + h) * SEQ + t) * HDIM + hd] =
                        (bf16)(val * sc);
                }
            }
        }
}

// ---------------- stage 3: causal flash attention ----------------
#define QBLK 128   // q rows per block (8 waves x 16)
#define KTIL 64    // keys per iteration

__global__ __launch_bounds__(256) void flash_attn(const bf16* __restrict__ Q,
                                                  const bf16* __restrict__ Km,
                                                  const bf16* __restrict__ Vm,
                                                  bf16* __restrict__ Out) {
    __shared__ bf16 lVt[HDIM][KTIL];       // V transposed [hd][key], 8 KB
    __shared__ bf16 lP[8][16][KTIL];       // per-wave P strips, 16 KB

    const unsigned QB = SEQ / QBLK;        // 32
    const unsigned qb = blockIdx.x & (QB - 1);
    const unsigned bh = blockIdx.x / QB;
    const unsigned b  = bh >> 4, h = bh & 15u;   // NHEAD = 16

    const bf16* Qh = Q  + (size_t)bh * SEQ * HDIM;
    const bf16* Kh = Km + (size_t)bh * SEQ * HDIM;
    const bf16* Vh = Vm + (size_t)bh * SEQ * HDIM;

    const int tid = threadIdx.x, wave = tid >> 5, lane = tid & 31;
    const unsigned col = lane & 15, half = lane >> 4;
    const unsigned q0  = qb * QBLK;
    const unsigned qw  = q0 + wave * 16;   // this wave's 16 q rows

    // Q strip fragments (head-dim 0..31 and 32..63); Q already carries 1/sqrt(hd)
    bf16x16 qf[2];
    qf[0] = load_fragA(Qh + (size_t)qw * HDIM + 0,  HDIM);
    qf[1] = load_fragA(Qh + (size_t)qw * HDIM + 32, HDIM);

    unsigned qrow[8];
    float mrow[8], lrow[8];
    f32x8 o[4];
    #pragma unroll
    for (int r = 0; r < 8; ++r) {
        qrow[r] = qw + r + 8 * half;
        mrow[r] = -1e30f;
        lrow[r] = 0.0f;
    }
    #pragma unroll
    for (int j = 0; j < 4; ++j) o[j] = zero8();

    const unsigned kend = q0 + QBLK;       // causal upper bound for this block
    for (unsigned kb = 0; kb < kend; kb += KTIL) {
        // stage V tile transposed: lVt[hd][key]
        #pragma unroll
        for (int i = 0; i < 16; ++i) {
            int linear = i * 256 + tid;
            int key = linear >> 6, hd = linear & 63;
            lVt[hd][key] = Vh[(size_t)(kb + key) * HDIM + hd];
        }
        __syncthreads();

        // S = Q * K^T  (K rows [key][hd] are exactly the B-operand stripe)
        f32x8 s[4];
        #pragma unroll
        for (int j = 0; j < 4; ++j) {
            s[j] = zero8();
            bf16x16 kf = load_fragB(Kh + (size_t)(kb + j * 16) * HDIM + 0, HDIM);
            s[j] = wmma_bf16(qf[0], kf, s[j]);
            kf = load_fragB(Kh + (size_t)(kb + j * 16) * HDIM + 32, HDIM);
            s[j] = wmma_bf16(qf[1], kf, s[j]);
        }

        // causal mask + running row max
        float rmax[8];
        #pragma unroll
        for (int r = 0; r < 8; ++r) rmax[r] = -1e30f;
        #pragma unroll
        for (int j = 0; j < 4; ++j) {
            unsigned key = kb + j * 16 + col;
            #pragma unroll
            for (int r = 0; r < 8; ++r) {
                float x = s[j][r];
                x = (key <= qrow[r]) ? x : -1e30f;
                s[j][r] = x;
                rmax[r] = fmaxf(rmax[r], x);
            }
        }
        #pragma unroll
        for (int off = 1; off < 16; off <<= 1)
            #pragma unroll
            for (int r = 0; r < 8; ++r)
                rmax[r] = fmaxf(rmax[r], __shfl_xor(rmax[r], off, 32));

        // online softmax update; write P (bf16) into this wave's LDS strip
        float corr[8], rsum[8];
        #pragma unroll
        for (int r = 0; r < 8; ++r) {
            float nm = fmaxf(mrow[r], rmax[r]);
            corr[r] = __expf(mrow[r] - nm);
            mrow[r] = nm;
            rsum[r] = 0.0f;
        }
        #pragma unroll
        for (int j = 0; j < 4; ++j)
            #pragma unroll
            for (int r = 0; r < 8; ++r) {
                float p = __expf(s[j][r] - mrow[r]);
                rsum[r] += p;
                lP[wave][r + 8 * half][j * 16 + col] = (bf16)p;
            }
        #pragma unroll
        for (int off = 1; off < 16; off <<= 1)
            #pragma unroll
            for (int r = 0; r < 8; ++r)
                rsum[r] += __shfl_xor(rsum[r], off, 32);
        #pragma unroll
        for (int r = 0; r < 8; ++r) lrow[r] = lrow[r] * corr[r] + rsum[r];
        #pragma unroll
        for (int j = 0; j < 4; ++j)
            #pragma unroll
            for (int r = 0; r < 8; ++r) o[j][r] *= corr[r];

        // O += P * V   (P re-striped via LDS; Vt rows are B-operand stripes)
        bf16x16 pf[2];
        pf[0] = load_fragA(&lP[wave][0][0], KTIL);        // keys 0..31
        pf[1] = load_fragA(&lP[wave][0][0] + 32, KTIL);   // keys 32..63
        #pragma unroll
        for (int j = 0; j < 4; ++j) {
            bf16x16 vf = load_fragB(&lVt[j * 16][0], KTIL);
            o[j] = wmma_bf16(pf[0], vf, o[j]);
            vf = load_fragB(&lVt[j * 16][0] + 32, KTIL);
            o[j] = wmma_bf16(pf[1], vf, o[j]);
        }
        __syncthreads();   // protect lVt before next stage overwrites it
    }

    // normalize and write to [B,T,DMODEL] bf16
    #pragma unroll
    for (int r = 0; r < 8; ++r) lrow[r] = 1.0f / lrow[r];
    #pragma unroll
    for (int j = 0; j < 4; ++j)
        #pragma unroll
        for (int r = 0; r < 8; ++r)
            Out[((size_t)b * SEQ + qrow[r]) * DMODEL + h * HDIM + j * 16 + col] =
                (bf16)(o[j][r] * lrow[r]);
}

// ---------------- host launcher ----------------
extern "C" void kernel_launch(void* const* d_in, const int* in_sizes, int n_in,
                              void* d_out, int out_size, void* d_ws, size_t ws_size,
                              hipStream_t stream) {
    const float* x      = (const float*)d_in[0];
    const float* w_qkv  = (const float*)d_in[1];
    const float* w_proj = (const float*)d_in[2];
    float* out = (float*)d_out;

    char* ws = (char*)d_ws;
    size_t off = 0;
    bf16* xb     = (bf16*)(ws + off); off += (size_t)MTOT * DMODEL * 2;   // 32 MB
    bf16* wqkvb  = (bf16*)(ws + off); off += (size_t)DMODEL * NQKV * 2;   //  6 MB
    bf16* qbuf   = (bf16*)(ws + off); off += (size_t)MTOT * DMODEL * 2;   // 32 MB
    bf16* kbuf   = (bf16*)(ws + off); off += (size_t)MTOT * DMODEL * 2;   // 32 MB
    bf16* vbuf   = (bf16*)(ws + off); off += (size_t)MTOT * DMODEL * 2;   // 32 MB
    bf16* attnb  = (bf16*)(ws + off); off += (size_t)MTOT * DMODEL * 2;   // 32 MB
    bf16* wprojb = (bf16*)(ws + off); off += (size_t)DMODEL * DMODEL * 2; //  2 MB

    // stage 1: conversions
    cvt_f32_bf16<<<2048, 256, 0, stream>>>(x, xb, (long long)MTOT * DMODEL);
    cvt_f32_bf16<<<1024, 256, 0, stream>>>(w_qkv, wqkvb, (long long)DMODEL * NQKV);
    cvt_f32_bf16<<<512, 256, 0, stream>>>(w_proj, wprojb, (long long)DMODEL * DMODEL);

    // stage 2: QKV projection (scatter into [B,H,T,HDIM], Q pre-scaled)
    {
        dim3 grid((MTOT / MBLK) * (NQKV / NBLK));   // 128 * 24 = 3072
        gemm_bf16<<<grid, 256, 0, stream>>>(xb, wqkvb, MTOT, NQKV, DMODEL, 1,
                                            nullptr, qbuf, kbuf, vbuf);
    }

    // stage 3: causal flash attention -> attnb [B,T,DMODEL]
    {
        dim3 grid(BATCH * NHEAD * (SEQ / QBLK));    // 64 * 32 = 2048
        flash_attn<<<grid, 256, 0, stream>>>(qbuf, kbuf, vbuf, attnb);
    }

    // stage 4: output projection -> f32 d_out
    {
        dim3 grid((MTOT / MBLK) * (DMODEL / NBLK)); // 128 * 8 = 1024
        gemm_bf16<<<grid, 256, 0, stream>>>(attnb, wprojb, MTOT, DMODEL, DMODEL, 0,
                                            out, nullptr, nullptr, nullptr);
    }
}
